// MultiHeadAttention_79577154060910
// MI455X (gfx1250) — matrix operand order repaired and checked by reading, not compiled, not run on hardware
//
#include <hip/hip_runtime.h>

// ---------------------------------------------------------------------------
// MI455X (gfx1250) multi-head attention, faithful to the "global softmax"
// reference. bf16 WMMA (16x16x32) with f32 accumulation everywhere.
//
// Shapes: B=2, S=2048, D=1024, H=16, HD=64.  Head bh = b*16+h is the
// contiguous 2048x64 block at offset bh*131072 of the projected tensors.
// ---------------------------------------------------------------------------

typedef __attribute__((ext_vector_type(8)))  __bf16 v8bf;
typedef __attribute__((ext_vector_type(16))) __bf16 v16bf;
typedef __attribute__((ext_vector_type(8)))  float  v8f;
typedef __attribute__((__vector_size__(16))) int    v4i;

#define S_SEQ   2048
#define D_MODEL 1024
#define HEADBLK 131072      // 2048*64 elements per (b,h) head block

// ---- CDNA5 async global->LDS path (guarded; falls back to sync staging) ----
#if __has_builtin(__builtin_amdgcn_global_load_async_to_lds_b128)
#define HAVE_ASYNC_LDS 1
typedef __attribute__((address_space(1))) v4i* as1_v4i;
typedef __attribute__((address_space(3))) v4i* as3_v4i;
#if __has_builtin(__builtin_amdgcn_s_wait_asynccnt)
#define WAIT_ASYNC(n) __builtin_amdgcn_s_wait_asynccnt(n)
#else
#define WAIT_ASYNC(n) asm volatile("s_wait_asynccnt %0" ::"n"(n) : "memory")
#endif
#else
#define HAVE_ASYNC_LDS 0
#define WAIT_ASYNC(n) ((void)0)
#endif

// Copy 32 bytes (16 bf16) global -> LDS.
__device__ __forceinline__ void copy32_to_lds(const __bf16* g, __bf16* l) {
#if HAVE_ASYNC_LDS
  __builtin_amdgcn_global_load_async_to_lds_b128(
      (as1_v4i)(uintptr_t)g, (as3_v4i)(uint32_t)(uintptr_t)l, 0, 0);
  __builtin_amdgcn_global_load_async_to_lds_b128(
      (as1_v4i)(uintptr_t)(g + 8), (as3_v4i)(uint32_t)(uintptr_t)(l + 8), 0, 0);
#else
  v8bf a = *(const v8bf*)g;
  v8bf b = *(const v8bf*)(g + 8);
  *(v8bf*)l = a;
  *(v8bf*)(l + 8) = b;
#endif
}

__device__ __forceinline__ v8f wmma_bf16(v16bf a, v16bf b, v8f c) {
  return __builtin_amdgcn_wmma_f32_16x16x32_bf16(
      /*neg_a=*/false, a, /*neg_b=*/false, b,
      /*c_mod=*/(short)0, c, /*reuse_a=*/false, /*reuse_b=*/false);
}

// A-operand (16x32 bf16): lane (m = lane&15, half = lane>>4) holds
// row[half*16 .. +15].
__device__ __forceinline__ v16bf load_a(const __bf16* row, int half) {
  v8bf lo = *(const v8bf*)(row + half * 16);
  v8bf hi = *(const v8bf*)(row + half * 16 + 8);
  return __builtin_shufflevector(lo, hi, 0,1,2,3,4,5,6,7,8,9,10,11,12,13,14,15);
}

// ---------------------------------------------------------------------------
// 1) Transpose + convert the three 1024x1024 f32 weight matrices to bf16.
//    Wt[z][n*1024 + k] = W[z][k*1024 + n]
// ---------------------------------------------------------------------------
__global__ void k_wtrans(const float* __restrict__ W0,
                         const float* __restrict__ W1,
                         const float* __restrict__ W2,
                         __bf16* __restrict__ Wt) {
  const float* W = (blockIdx.z == 0) ? W0 : (blockIdx.z == 1) ? W1 : W2;
  __bf16* out = Wt + (size_t)blockIdx.z * (D_MODEL * D_MODEL);
  int tid = blockIdx.x * 256 + threadIdx.x;      // 0 .. 1M-1
  int n = tid >> 10, k = tid & 1023;
  out[tid] = (__bf16)W[k * D_MODEL + n];
}

// ---------------------------------------------------------------------------
// 2) Projection GEMM: P[z] = X[z] @ W[z], output bf16 (Q pre-scaled by 1/8).
//    One wave computes a 16x128 strip (8 WMMAs per A-load) to halve L2
//    traffic; grid (64, 8, 3), block (32,4).
// ---------------------------------------------------------------------------
__global__ __launch_bounds__(128) void k_proj(const float* __restrict__ q,
                                              const float* __restrict__ k,
                                              const float* __restrict__ v,
                                              const __bf16* __restrict__ Wt,
                                              __bf16* __restrict__ P) {
  int z = blockIdx.z;
  const float*  X   = (z == 0) ? q : (z == 1) ? k : v;
  const __bf16* W   = Wt + (size_t)z * (D_MODEL * D_MODEL);
  __bf16*       out = P  + (size_t)z * (4096 * D_MODEL);
  float scale = (z == 0) ? 0.125f : 1.0f;        // fold 1/sqrt(64) into Q

  int lane = threadIdx.x, half = lane >> 4, lr = lane & 15;
  int m0 = (blockIdx.x * 4 + threadIdx.y) * 16;  // 0..4080
  int n0 = blockIdx.y * 128;                     // 0..896

  const float* xrow = X + (size_t)(m0 + lr) * D_MODEL;
  v8f acc[8];
#pragma unroll
  for (int j = 0; j < 8; ++j) acc[j] = (v8f){0,0,0,0,0,0,0,0};

  for (int kk = 0; kk < D_MODEL; kk += 32) {
    __builtin_prefetch(xrow + kk + 64, 0, 1);
    v8f f0 = *(const v8f*)(xrow + kk + half * 16);
    v8f f1 = *(const v8f*)(xrow + kk + half * 16 + 8);
    v16bf a;
#pragma unroll
    for (int i = 0; i < 8; ++i) { a[i] = (__bf16)f0[i]; a[8 + i] = (__bf16)f1[i]; }
    const __bf16* wb = W + (size_t)(n0 + lr) * D_MODEL + kk + half * 16;
#pragma unroll
    for (int j = 0; j < 8; ++j)
      acc[j] = wmma_bf16(a, *(const v16bf*)(wb + (size_t)(16 * j) * D_MODEL), acc[j]);
  }
#pragma unroll
  for (int r = 0; r < 8; ++r) {
    size_t row = (size_t)(m0 + r + 8 * half) * D_MODEL + n0 + lr;
#pragma unroll
    for (int j = 0; j < 8; ++j)
      out[row + 16 * j] = (__bf16)(acc[j][r] * scale);
  }
}

// ---------------------------------------------------------------------------
// 3) Per-head transpose of V: Vt[bh][hd][s] = Vp[bh][s][hd]  (bf16)
// ---------------------------------------------------------------------------
__global__ void k_vtrans(const __bf16* __restrict__ P, __bf16* __restrict__ Vt) {
  const __bf16* Vp = P + (size_t)2 * 4096 * D_MODEL;
  int tid = blockIdx.x * 256 + threadIdx.x;      // 0 .. 4194303
  int s  = tid & 2047;
  int n  = (tid >> 11) & 63;
  int bh = tid >> 17;
  Vt[tid] = Vp[(size_t)bh * HEADBLK + (size_t)s * 64 + n];
}

// ---------------------------------------------------------------------------
// 4) Pass 1: scores = (Q/8)K^T per head; online (max,sum) over ALL elements
//    plus running max over strict-upper elements. One wave handles a 16x256
//    strip; partial (m, s, maxU) written per wave. Grid (256,32), block (32,4).
// ---------------------------------------------------------------------------
__global__ __launch_bounds__(128) void k_pass1(const __bf16* __restrict__ P,
                                               float4* __restrict__ part) {
  int lane = threadIdx.x, half = lane >> 4, lr = lane & 15;
  int wl = blockIdx.x * 4 + threadIdx.y;         // 0..1023
  int bh = blockIdx.y;
  int qt = wl >> 3, ns = wl & 7;
  int q0 = qt * 16, n0b = ns * 256;

  const __bf16* Qh = P + (size_t)bh * HEADBLK;
  const __bf16* Kh = P + (size_t)4096 * D_MODEL + (size_t)bh * HEADBLK;
  const __bf16* qrow = Qh + (size_t)(q0 + lr) * 64;
  v16bf a0 = load_a(qrow, half), a1 = load_a(qrow + 32, half);

  float m = -3.0e38f, sum = 0.0f, mU = -3.0e38f;
  for (int t = 0; t < 16; ++t) {
    int n0 = n0b + t * 16;
    const __bf16* krow = Kh + (size_t)(n0 + lr) * 64;
    __builtin_prefetch(krow + 16 * 64, 0, 1);
    v8f c = {0,0,0,0,0,0,0,0};
    c = wmma_bf16(a0, *(const v16bf*)(krow + half * 16),      c);
    c = wmma_bf16(a1, *(const v16bf*)(krow + 32 + half * 16), c);
    int kcol = n0 + lr;
#pragma unroll
    for (int r = 0; r < 8; ++r) {
      float vv = c[r];
      float nm = fmaxf(m, vv);
      sum = sum * __expf(m - nm) + __expf(vv - nm);
      m = nm;
      if (kcol > q0 + r + 8 * half) mU = fmaxf(mU, vv);
    }
  }
  // wave32 butterfly combine
#pragma unroll
  for (int off = 16; off >= 1; off >>= 1) {
    float om = __shfl_xor(m, off, 32);
    float os = __shfl_xor(sum, off, 32);
    float ou = __shfl_xor(mU, off, 32);
    float nm = fmaxf(m, om);
    sum = sum * __expf(m - nm) + os * __expf(om - nm);
    m = nm; mU = fmaxf(mU, ou);
  }
  if (lane == 0) part[(size_t)bh * 1024 + wl] = make_float4(m, sum, mU, 0.0f);
}

// 4b) Deterministic combine: -> scal[0]=max1, scal[1]=1/sum1, scal[2]=maxU_att
__global__ void k_comb1(const float4* __restrict__ part, float* __restrict__ scal) {
  __shared__ float sm[256], ss[256], su[256];
  int t = threadIdx.x;
  float m = -3.0e38f, s = 0.0f, u = -3.0e38f;
  for (int i = t; i < 32768; i += 256) {
    float4 p = part[i];
    float nm = fmaxf(m, p.x);
    s = s * __expf(m - nm) + p.y * __expf(p.x - nm);
    m = nm; u = fmaxf(u, p.z);
  }
  sm[t] = m; ss[t] = s; su[t] = u; __syncthreads();
  for (int off = 128; off >= 1; off >>= 1) {
    if (t < off) {
      float om = sm[t + off], os = ss[t + off];
      float nm = fmaxf(sm[t], om);
      ss[t] = ss[t] * __expf(sm[t] - nm) + os * __expf(om - nm);
      sm[t] = nm; su[t] = fmaxf(su[t], su[t + off]);
    }
    __syncthreads();
  }
  if (t == 0) {
    float max1 = sm[0], sum1 = ss[0], maxU = su[0];
    scal[0] = max1;
    scal[1] = 1.0f / sum1;
    scal[2] = __expf(maxU - max1) / sum1;   // maxU_att: max of S_att on upper tri
  }
}

// ---------------------------------------------------------------------------
// 5) Pass 2: sum2 = sum over strict-upper of exp(S_att - maxU_att).
//    Lower-triangle terms are exactly 0 in f32 (exp(~-1e9)) -> skipped.
// ---------------------------------------------------------------------------
__global__ __launch_bounds__(128) void k_pass2(const __bf16* __restrict__ P,
                                               const float* __restrict__ scal,
                                               float* __restrict__ part2) {
  int lane = threadIdx.x, half = lane >> 4, lr = lane & 15;
  int wl = blockIdx.x * 4 + threadIdx.y;
  int bh = blockIdx.y;
  int qt = wl >> 3, ns = wl & 7;
  int q0 = qt * 16, n0b = ns * 256;
  size_t pidx = (size_t)bh * 1024 + wl;

  if (n0b + 255 <= q0) {                         // strip entirely lower
    if (lane == 0) part2[pidx] = 0.0f;
    return;
  }
  float max1 = scal[0], inv1 = scal[1], mUatt = scal[2];

  const __bf16* Qh = P + (size_t)bh * HEADBLK;
  const __bf16* Kh = P + (size_t)4096 * D_MODEL + (size_t)bh * HEADBLK;
  const __bf16* qrow = Qh + (size_t)(q0 + lr) * 64;
  v16bf a0 = load_a(qrow, half), a1 = load_a(qrow + 32, half);

  float acc = 0.0f;
  for (int t = 0; t < 16; ++t) {
    int n0 = n0b + t * 16;
    if (n0 + 15 <= q0) continue;                 // tile entirely lower
    const __bf16* krow = Kh + (size_t)(n0 + lr) * 64;
    v8f c = {0,0,0,0,0,0,0,0};
    c = wmma_bf16(a0, *(const v16bf*)(krow + half * 16),      c);
    c = wmma_bf16(a1, *(const v16bf*)(krow + 32 + half * 16), c);
    int kcol = n0 + lr;
#pragma unroll
    for (int r = 0; r < 8; ++r) {
      if (kcol > q0 + r + 8 * half) {
        float satt = __expf(c[r] - max1) * inv1;
        acc += __expf(satt - mUatt);
      }
    }
  }
#pragma unroll
  for (int off = 16; off >= 1; off >>= 1) acc += __shfl_xor(acc, off, 32);
  if (lane == 0) part2[pidx] = acc;
}

// 5b) scal[3] = 1/sum2
__global__ void k_comb2(const float* __restrict__ part2, float* __restrict__ scal) {
  __shared__ float sm[256];
  int t = threadIdx.x;
  float s = 0.0f;
  for (int i = t; i < 32768; i += 256) s += part2[i];
  sm[t] = s; __syncthreads();
  for (int off = 128; off >= 1; off >>= 1) {
    if (t < off) sm[t] += sm[t + off];
    __syncthreads();
  }
  if (t == 0) scal[3] = 1.0f / sm[0];
}

// ---------------------------------------------------------------------------
// 6) Output pass: O[bh] = A @ V[bh], A recomputed tile-by-tile.
//    A block of 4 waves shares (bh, 64 q rows); the per-step 4KB K tile and
//    4KB V^T tile are staged into LDS with double-buffered async
//    global->LDS copies (one step ahead), gated by s_wait_asynccnt +
//    workgroup barriers. Weight C-tiles convert to WMMA A-layout through a
//    per-wave LDS staging buffer (LDS is in-order per wave).
// ---------------------------------------------------------------------------
__global__ __launch_bounds__(128) void k_out(const __bf16* __restrict__ P,
                                             const __bf16* __restrict__ Vt,
                                             const float* __restrict__ scal,
                                             float* __restrict__ out) {
  __shared__ __bf16 kbuf[2][32 * 64];   // K rows kb..kb+31 (row-major, 64 wide)
  __shared__ __bf16 vbuf[2][64 * 32];   // V^T rows hd 0..63, cols kb..kb+31
  __shared__ __bf16 smw[4][16 * 32];    // per-wave weight staging

  int lane = threadIdx.x, half = lane >> 4, lr = lane & 15, ty = threadIdx.y;
  int tid = ty * 32 + lane;              // 0..127
  int qt = blockIdx.x * 4 + ty;          // 0..127
  int bh = blockIdx.y;
  int q0 = qt * 16;
  int q0base = blockIdx.x * 64;          // min q0 in this block

  float max1 = scal[0], inv1 = scal[1], mUatt = scal[2], inv2 = scal[3];

  const __bf16* Qh  = P + (size_t)bh * HEADBLK;
  const __bf16* Kh  = P + (size_t)4096 * D_MODEL + (size_t)bh * HEADBLK;
  const __bf16* Vth = Vt + (size_t)bh * (64 * S_SEQ);
  const __bf16* qrow = Qh + (size_t)(q0 + lr) * 64;
  v16bf a0 = load_a(qrow, half), a1 = load_a(qrow + 32, half);

  v8f acc0 = {0,0,0,0,0,0,0,0}, acc1 = acc0, acc2 = acc0, acc3 = acc0;
  __bf16* my = &smw[ty][0];

  // first kb with any k > q for the smallest q0 in the block
  int kb0 = (q0base >= 31) ? ((((q0base - 31) >> 5) << 5) + 32) : 0;

  // stage(kb, buf): 4 async b128 per thread (2 for K, 2 for V^T)
  int vr = tid >> 1, vh = tid & 1;
#define STAGE(KB, BUF)                                                        \
  do {                                                                        \
    copy32_to_lds(Kh + (size_t)(KB) * 64 + tid * 16, &kbuf[BUF][tid * 16]);   \
    copy32_to_lds(Vth + (size_t)vr * S_SEQ + (KB) + vh * 16,                  \
                  &vbuf[BUF][vr * 32 + vh * 16]);                             \
  } while (0)

  int cur = 0;
  STAGE(kb0, 0);
  for (int kb = kb0; kb < S_SEQ; kb += 32) {
    if (kb + 32 < S_SEQ) {
      STAGE(kb + 32, cur ^ 1);
      WAIT_ASYNC(4);                    // current buffer's 4 copies complete
    } else {
      WAIT_ASYNC(0);
    }
    __syncthreads();                    // staged data visible block-wide

    const __bf16* kr0 = &kbuf[cur][0] + lr * 64;
    const __bf16* kr1 = &kbuf[cur][0] + (16 + lr) * 64;
    v8f c0 = {0,0,0,0,0,0,0,0}, c1 = c0;
    c0 = wmma_bf16(a0, *(const v16bf*)(kr0 + half * 16),      c0);
    c0 = wmma_bf16(a1, *(const v16bf*)(kr0 + 32 + half * 16), c0);
    c1 = wmma_bf16(a0, *(const v16bf*)(kr1 + half * 16),      c1);
    c1 = wmma_bf16(a1, *(const v16bf*)(kr1 + 32 + half * 16), c1);

    // second-softmax weights -> bf16 -> per-wave LDS (row-major 16x32)
#pragma unroll
    for (int r = 0; r < 8; ++r) {
      int qrow_i = q0 + r + 8 * half;
      float w0 = (kb + lr > qrow_i)
                     ? __expf(__expf(c0[r] - max1) * inv1 - mUatt) * inv2 : 0.0f;
      float w1 = (kb + 16 + lr > qrow_i)
                     ? __expf(__expf(c1[r] - max1) * inv1 - mUatt) * inv2 : 0.0f;
      my[(r + 8 * half) * 32 + lr]      = (__bf16)w0;
      my[(r + 8 * half) * 32 + 16 + lr] = (__bf16)w1;
    }
    asm volatile("s_wait_dscnt 0x0" ::: "memory");   // LDS in-order per wave

    v8bf wlo = *(const v8bf*)(my + lr * 32 + half * 16);
    v8bf whi = *(const v8bf*)(my + lr * 32 + half * 16 + 8);
    v16bf aw = __builtin_shufflevector(wlo, whi,
                 0,1,2,3,4,5,6,7,8,9,10,11,12,13,14,15);
    asm volatile("" ::: "memory");

    const __bf16* vb = &vbuf[cur][0] + lr * 32 + half * 16;
    acc0 = wmma_bf16(aw, *(const v16bf*)(vb),            acc0);
    acc1 = wmma_bf16(aw, *(const v16bf*)(vb + 16 * 32),  acc1);
    acc2 = wmma_bf16(aw, *(const v16bf*)(vb + 32 * 32),  acc2);
    acc3 = wmma_bf16(aw, *(const v16bf*)(vb + 48 * 32),  acc3);

    __syncthreads();                    // all reads done before next overwrite
    cur ^= 1;
  }
#undef STAGE

  int b = bh >> 4, h = bh & 15;
  float* ob = out + (size_t)b * (S_SEQ * D_MODEL) + (size_t)h * 64;
#pragma unroll
  for (int r = 0; r < 8; ++r) {
    size_t row = (size_t)(q0 + r + 8 * half) * D_MODEL + lr;
    ob[row]      = acc0[r];
    ob[row + 16] = acc1[r];
    ob[row + 32] = acc2[r];
    ob[row + 48] = acc3[r];
  }
}

// ---------------------------------------------------------------------------
// Host-side orchestration (all on `stream`, graph-capture safe, deterministic)
// ---------------------------------------------------------------------------
extern "C" void kernel_launch(void* const* d_in, const int* in_sizes, int n_in,
                              void* d_out, int out_size, void* d_ws, size_t ws_size,
                              hipStream_t stream) {
  const float* q  = (const float*)d_in[0];
  const float* k  = (const float*)d_in[1];
  const float* v  = (const float*)d_in[2];
  const float* WQ = (const float*)d_in[3];
  const float* WK = (const float*)d_in[4];
  const float* WV = (const float*)d_in[5];
  float* out = (float*)d_out;

  char* ws = (char*)d_ws;
  __bf16* Wt = (__bf16*)(ws);                    //  6,291,456 B (3x 1024x1024 bf16, transposed)
  __bf16* P  = (__bf16*)(ws + 6291456);          // 25,165,824 B (Qp,Kp,Vp bf16; Q pre-scaled)
  __bf16* Vt = (__bf16*)(ws + 31457280);         //  8,388,608 B (per-head V transposed)
  float4* p1 = (float4*)(ws + 39845888);         //    524,288 B (pass1 partials)
  float*  p2 = (float*) (ws + 40370176);         //    131,072 B (pass2 partials)
  float*  sc = (float*) (ws + 40501248);         //         16 B (max1, 1/sum1, maxU_att, 1/sum2)

  k_wtrans<<<dim3(4096, 1, 3), 256, 0, stream>>>(WQ, WK, WV, Wt);
  k_proj  <<<dim3(64, 8, 3), dim3(32, 4), 0, stream>>>(q, k, v, Wt, P);
  k_vtrans<<<dim3(16384), 256, 0, stream>>>(P, Vt);
  k_pass1 <<<dim3(256, 32), dim3(32, 4), 0, stream>>>(P, p1);
  k_comb1 <<<1, 256, 0, stream>>>(p1, sc);
  k_pass2 <<<dim3(256, 32), dim3(32, 4), 0, stream>>>(P, sc, p2);
  k_comb2 <<<1, 256, 0, stream>>>(p2, sc);
  k_out   <<<dim3(32, 32), dim3(32, 4), 0, stream>>>(P, Vt, sc, out);
}
